// GT_Modular_Transformer_86827058856237
// MI455X (gfx1250) — compile-verified, matrix operand order checked by reading, not run on hardware
//
#include <hip/hip_runtime.h>
#include <hip/hip_bf16.h>
#include <stdint.h>

// ---------------------------------------------------------------------------
// Modular transformer layer for MI455X (gfx1250, wave32, WMMA 16x16x32 f16).
//   - weights f32 -> f16 pre-packed into WMMA B-fragment tile order
//   - A strips staged per-workgroup into LDS with async global->LDS b128
//     copies (ASYNCcnt) and consumed via ds_load_b128 fragments
//   - attention: QK^T WMMA -> LDS scores -> masked softmax -> P@V WMMA
//   - per-rule MLP with zero-padded K (160 / 416), r_scores mixing in f32
// ---------------------------------------------------------------------------

typedef __attribute__((ext_vector_type(16))) _Float16 v16h;
typedef __attribute__((ext_vector_type(8)))  _Float16 v8h;
typedef __attribute__((ext_vector_type(8)))  float    v8f;

#define WMMA_F32_16x16x32_F16(a, b, c) \
  __builtin_amdgcn_wmma_f32_16x16x32_f16(false, (a), false, (b), (short)0, (c), false, false)

// ---- problem constants ----------------------------------------------------
#define BN    4096       // b*n rows
#define DIM_  512
#define G_    32         // R*H groups
#define HD_   16
#define R_    4
#define H_    8
#define INTER 405
#define K1P   160        // 144 padded
#define K2P   416        // 405 padded
#define HROW  (R_*K1P)   // 640 halfs / row of concat buffer
#define MROW  (R_*K2P)   // 1664 halfs / row of hmid buffer
#define SCALE_Q 0.25f    // HD^-0.5
#define TILEH 512        // halves per packed 32x16 B tile (32 lanes * 16)

// ---- gfx1250 async global->LDS copy (16B per lane) ------------------------
__device__ __forceinline__ void async_g2l_b128(void* lds, const void* g) {
  const unsigned int loff = (unsigned int)(uintptr_t)lds;     // LDS = addr[31:0]
  const unsigned long long ga = (unsigned long long)(uintptr_t)g;
  asm volatile("global_load_async_to_lds_b128 %0, %1, off"
               :: "v"(loff), "v"(ga) : "memory");
}
__device__ __forceinline__ void wait_async0() {
#if __has_builtin(__builtin_amdgcn_s_wait_asynccnt)
  __builtin_amdgcn_s_wait_asynccnt(0);
#else
  asm volatile("s_wait_asynccnt 0" ::: "memory");
#endif
}

// Stage a 16-row A strip [16][rowHalfs] (rowHalfs multiple of 8) into LDS.
// Each of 8 waves copies 2 rows; lanes stride 16B chunks. Waits on ASYNCcnt, then barriers.
__device__ __forceinline__ void stage_strip(_Float16* lds, const _Float16* g,
                                            int rowHalfs, int ldg) {
  const int wave = threadIdx.x >> 5;
  const int lane = threadIdx.x & 31;
  const int cpr = rowHalfs >> 3;               // 16B chunks per row
#pragma unroll
  for (int rr = 0; rr < 2; ++rr) {
    const int row = wave * 2 + rr;
    for (int c = lane; c < cpr; c += 32)
      async_g2l_b128(lds + row * rowHalfs + c * 8,
                     g + (size_t)row * ldg + c * 8);
  }
  wait_async0();
  __syncthreads();
}

// ---- fragment helpers -----------------------------------------------------
__device__ __forceinline__ v16h frag_cat(v8h lo, v8h hi) {
  return __builtin_shufflevector(lo, hi, 0,1,2,3,4,5,6,7,8,9,10,11,12,13,14,15);
}

// A 16x32 fragment, all K valid: two contiguous b128 loads per lane.
__device__ __forceinline__ v16h lda_full(const _Float16* Arow0, int lda, int k0) {
  const int lane = threadIdx.x & 31;
  const _Float16* p = Arow0 + (size_t)(lane & 15) * lda + k0 + 8 * (lane >> 4);
  return frag_cat(*(const v8h*)p, *(const v8h*)(p + 16));
}

// A 16x32 fragment with only K=0..15 valid (HD=16 zero-padded to 32).
__device__ __forceinline__ v16h lda_half16(const _Float16* __restrict__ Arow0,
                                           int lda) {
  const int lane = threadIdx.x & 31;
  const _Float16* p = Arow0 + (size_t)(lane & 15) * lda + 8 * (lane >> 4);
  v8h z = {};
  return frag_cat(*(const v8h*)p, z);
}

// Packed B tile at pointer p (pre-offset by lane*16).
__device__ __forceinline__ v16h ldb_ptr(const _Float16* __restrict__ p) {
  return frag_cat(*(const v8h*)p, *(const v8h*)(p + 8));
}

// ---- utility kernels ------------------------------------------------------
__global__ void cvt_f32_f16_kernel(const float* __restrict__ src,
                                   _Float16* __restrict__ dst, int count) {
  int i = blockIdx.x * blockDim.x + threadIdx.x;
  const int stride = gridDim.x * blockDim.x;
  for (; i < count; i += stride) dst[i] = (_Float16)src[i];
}

__global__ void zero_f16_kernel(_Float16* __restrict__ dst, int count) {
  int i = blockIdx.x * blockDim.x + threadIdx.x;
  const int stride = gridDim.x * blockDim.x;
  for (; i < count; i += stride) dst[i] = (_Float16)0.f;
}

// ---- weight pack: f32 [K][N] -> f16 WMMA B-fragment tiles -----------------
__global__ __launch_bounds__(32)
void pack_b_f32_kernel(const float* __restrict__ src, _Float16* __restrict__ dst,
                       int K, int N, int Kt) {
  const int tile = blockIdx.x;
  const int kt = tile % Kt, nt = tile / Kt;
  const int lane = threadIdx.x & 31;
  const int n = lane & 15, hi = lane >> 4;
  const int col = nt * 16 + n;
  const int colc = (col < N) ? col : 0;
  const bool cok = (col < N);
  _Float16* out = dst + (size_t)tile * TILEH + lane * 16;
#pragma unroll
  for (int j = 0; j < 4; ++j) {
    const int k0 = kt * 32 + 8 * hi + 2 * j;
#pragma unroll
    for (int s = 0; s < 2; ++s) {
      const int ka = k0 + 16 * s;
      const int kb = ka + 1;
      const int kac = (ka < K) ? ka : 0;
      const int kbc = (kb < K) ? kb : 0;
      const float va = src[(size_t)kac * N + colc];
      const float vb = src[(size_t)kbc * N + colc];
      out[8 * s + 2 * j]     = (cok && ka < K) ? (_Float16)va : (_Float16)0.f;
      out[8 * s + 2 * j + 1] = (cok && kb < K) ? (_Float16)vb : (_Float16)0.f;
    }
  }
}

// ---- repack v (f16 [4096][512]) into per-(b,g,jt) B tiles -----------------
__global__ __launch_bounds__(32)
void pack_v_kernel(const _Float16* __restrict__ vh, _Float16* __restrict__ vpack) {
  const int t = blockIdx.x;
  const int jt = t & 15;
  const int g  = (t >> 4) & 31;
  const int bb = t >> 9;
  const int lane = threadIdx.x & 31;
  const int n = lane & 15, hi = lane >> 4;
  const _Float16* base = vh + ((size_t)bb * 512 + jt * 32) * DIM_ + g * HD_ + n;
  _Float16* out = vpack + (size_t)t * TILEH + lane * 16;
#pragma unroll
  for (int j = 0; j < 4; ++j) {
    const int k = 8 * hi + 2 * j;
#pragma unroll
    for (int s = 0; s < 2; ++s) {
      out[8 * s + 2 * j]     = base[(size_t)(k + 16 * s)     * DIM_];
      out[8 * s + 2 * j + 1] = base[(size_t)(k + 16 * s + 1) * DIM_];
    }
  }
}

// ---- projection GEMM: C[M][N] = (A[M][K] @ B + bias) * scale --------------
// A strip staged in LDS (async), B pre-packed (pointer-increment per K-step).
__global__ __launch_bounds__(256)
void gemm_bias_f16_kernel(const _Float16* __restrict__ A,
                          const _Float16* __restrict__ Bp,
                          const float* __restrict__ bias,
                          _Float16* __restrict__ C,
                          int N, int K, float scale) {
  __shared__ __attribute__((aligned(16))) _Float16 As[16 * DIM_];
  const int lane = threadIdx.x & 31;
  const int wave = threadIdx.x >> 5;
  const int row0 = blockIdx.x * 16;
  const int maxnt = (N >> 4) - 1;
  int ntile = blockIdx.y * 8 + wave;
  const bool active = (ntile <= maxnt);
  if (!active) ntile = maxnt;
  const int Kt = K >> 5;

  stage_strip(As, A + (size_t)row0 * K, K, K);

  const int n = lane & 15, hi = lane >> 4;
  const _Float16* bp = Bp + (size_t)ntile * Kt * TILEH + lane * 16;
  v8f acc = {};
  for (int kt = 0; kt < Kt; ++kt) {
    v16h a = lda_full(As, K, kt * 32);
    v16h b = ldb_ptr(bp);
    bp += TILEH;
    acc = WMMA_F32_16x16x32_F16(a, b, acc);
  }
  if (active) {
    const int col0 = ntile * 16;
    const float bc = bias[col0 + n];
#pragma unroll
    for (int i = 0; i < 8; ++i) {
      const int row = row0 + i + 8 * hi;
      C[(size_t)row * N + col0 + n] = (_Float16)((acc[i] + bc) * scale);
    }
  }
}

// ---- in-projection: x @ Win + b_in -> h[bn][r][128..143] ------------------
__global__ __launch_bounds__(256)
void inproj_kernel(const _Float16* __restrict__ xh,
                   const _Float16* __restrict__ winp,
                   const float* __restrict__ b_in,
                   _Float16* __restrict__ hbuf) {
  __shared__ __attribute__((aligned(16))) _Float16 As[16 * DIM_];
  const int lane = threadIdx.x & 31;
  const int wave = threadIdx.x >> 5;
  const int row0 = blockIdx.x * 16;
  const bool active = (wave < 4);
  const int nt = active ? wave : 3;           // rule index
  const int Kt = DIM_ >> 5;                   // 16

  stage_strip(As, xh + (size_t)row0 * DIM_, DIM_, DIM_);

  const int n = lane & 15, hi = lane >> 4;
  const _Float16* bp = winp + (size_t)nt * Kt * TILEH + lane * 16;
  v8f acc = {};
  for (int kt = 0; kt < Kt; ++kt) {
    v16h a = lda_full(As, DIM_, kt * 32);
    v16h b = ldb_ptr(bp);
    bp += TILEH;
    acc = WMMA_F32_16x16x32_F16(a, b, acc);
  }
  if (active) {
    const float bc = b_in[nt * 16 + n];
#pragma unroll
    for (int i = 0; i < 8; ++i) {
      const int row = row0 + i + 8 * hi;
      hbuf[(size_t)row * HROW + nt * K1P + 128 + n] = (_Float16)(acc[i] + bc);
    }
  }
}

// ---- attention: per (batch,group), 16 query rows per workgroup ------------
__global__ __launch_bounds__(32)
void attention_kernel(const _Float16* __restrict__ qh,
                      const _Float16* __restrict__ kh,
                      const _Float16* __restrict__ vpack,
                      _Float16* __restrict__ hbuf) {
  __shared__ __attribute__((aligned(16))) float    S[16][512];
  __shared__ __attribute__((aligned(16))) _Float16 P[16][512];

  const int bg = blockIdx.x;        // 0..255
  const int bb = bg >> 5;           // batch
  const int g  = bg & 31;           // group (rule-major: r*8 + h)
  const int rr = g >> 3, hh = g & 7;
  const int it = blockIdx.y;        // query row tile, 0..31
  const int lane = threadIdx.x & 31;
  const int n = lane & 15, hi = lane >> 4;
  const float NEG_INF = -__builtin_inff();

  const _Float16* qbase = qh + ((size_t)(bb * 512 + it * 16)) * DIM_ + g * HD_;
  const v16h aq = lda_half16(qbase, DIM_);

  // ---- S = Q K^T (B fragment of K^T == A-half layout of K rows) ----
  for (int jt = 0; jt < 32; ++jt) {
    const _Float16* kbase = kh + ((size_t)(bb * 512 + jt * 16)) * DIM_ + g * HD_;
    v16h bk = lda_half16(kbase, DIM_);
    v8f c = {};
    v8f s = WMMA_F32_16x16x32_F16(aq, bk, c);
#pragma unroll
    for (int i = 0; i < 8; ++i) S[i + 8 * hi][jt * 16 + n] = s[i];
  }
  __syncthreads();

  // ---- masked softmax over 512 keys; lanes (l, l+16) share row l%16 ----
  {
    const int row = lane & 15;
    const int par = lane >> 4;
    const int gi = it * 16 + row;
    float mx = NEG_INF;
    for (int c = par; c < 512; c += 2) {
      const float v = (c == gi) ? NEG_INF : S[row][c];
      mx = fmaxf(mx, v);
    }
    mx = fmaxf(mx, __shfl_xor(mx, 16, 32));
    float sum = 0.f;
    for (int c = par; c < 512; c += 2) {
      const float v = (c == gi) ? NEG_INF : S[row][c];
      const float e = __expf(v - mx);
      S[row][c] = e;
      sum += e;
    }
    sum += __shfl_xor(sum, 16, 32);
    const float inv = 1.f / sum;
    for (int c = par; c < 512; c += 2)
      P[row][c] = (_Float16)(S[row][c] * inv);
  }
  __syncthreads();

  // ---- out = P @ V over 512-key contraction (V pre-packed) ----
  const _Float16* vp = vpack + ((size_t)(bb * 32 + g) * 16) * TILEH + lane * 16;
  v8f acc = {};
  for (int jt = 0; jt < 16; ++jt) {
    v16h ap = lda_full(&P[0][0], 512, jt * 32);
    v16h bv = ldb_ptr(vp);
    vp += TILEH;
    acc = WMMA_F32_16x16x32_F16(ap, bv, acc);
  }
#pragma unroll
  for (int i = 0; i < 8; ++i) {
    const int row = i + 8 * hi;
    const size_t bn = (size_t)bb * 512 + it * 16 + row;
    hbuf[bn * HROW + rr * K1P + hh * HD_ + n] = (_Float16)acc[i];
  }
}

// ---- MLP1 per rule: relu(h @ W1 + b1) -> hmid (K2P padded) ----------------
__global__ __launch_bounds__(256)
void mlp1_kernel(const _Float16* __restrict__ hbuf,
                 const _Float16* __restrict__ w1p,
                 const float* __restrict__ b1,
                 _Float16* __restrict__ hmid) {
  __shared__ __attribute__((aligned(16))) _Float16 As[16 * K1P];
  const int lane = threadIdx.x & 31;
  const int wave = threadIdx.x >> 5;
  const int r = blockIdx.z;
  const int row0 = blockIdx.x * 16;
  int ntile = blockIdx.y * 8 + wave;          // 26 tiles cover 416 cols
  const bool active = (ntile < 26);
  if (!active) ntile = 25;
  const int Kt = K1P >> 5;                    // 5

  stage_strip(As, hbuf + (size_t)row0 * HROW + r * K1P, K1P, HROW);

  const int n = lane & 15, hi = lane >> 4;
  const _Float16* bp = w1p + ((size_t)r * 26 + ntile) * Kt * TILEH + lane * 16;
  v8f acc = {};
#pragma unroll
  for (int kt = 0; kt < Kt; ++kt) {
    v16h a = lda_full(As, K1P, kt * 32);
    v16h b = ldb_ptr(bp);
    bp += TILEH;
    acc = WMMA_F32_16x16x32_F16(a, b, acc);
  }
  if (active) {
    const int col = ntile * 16 + n;
    const float bc = (col < INTER) ? b1[r * INTER + col] : 0.f;
#pragma unroll
    for (int i = 0; i < 8; ++i) {
      const int row = row0 + i + 8 * hi;
      const float v = (col < INTER) ? fmaxf(acc[i] + bc, 0.f) : 0.f;
      hmid[(size_t)row * MROW + r * K2P + ntile * 16 + n] = (_Float16)v;
    }
  }
}

// ---- MLP2 + rule mixing: y = sum_r (hmid_r @ W2_r + b2_r) * r_scores ------
__global__ __launch_bounds__(256)
void mlp2_kernel(const _Float16* __restrict__ hmid,
                 const _Float16* __restrict__ w2p,
                 const float* __restrict__ b2,
                 const float* __restrict__ rsc,   // [4096][4]
                 float* __restrict__ y) {
  __shared__ __attribute__((aligned(16))) _Float16 As[16 * K2P];
  const int lane = threadIdx.x & 31;
  const int wave = threadIdx.x >> 5;
  const int row0 = blockIdx.x * 16;
  const int ntile = blockIdx.y * 8 + wave;    // 32 tiles, exact
  const int n = lane & 15, hi = lane >> 4;
  const int Kt = K2P >> 5;                    // 13

  v8f ysum = {};
  for (int r = 0; r < R_; ++r) {
    if (r) __syncthreads();                   // WAR on As
    stage_strip(As, hmid + (size_t)row0 * MROW + r * K2P, K2P, MROW);

    const _Float16* bp = w2p + ((size_t)r * 32 + ntile) * Kt * TILEH + lane * 16;
    v8f acc = {};
    for (int kt = 0; kt < Kt; ++kt) {
      v16h a = lda_full(As, K2P, kt * 32);
      v16h b = ldb_ptr(bp);
      bp += TILEH;
      acc = WMMA_F32_16x16x32_F16(a, b, acc);
    }
    const float bc = b2[r * DIM_ + ntile * 16 + n];
#pragma unroll
    for (int i = 0; i < 8; ++i) {
      const int row = row0 + i + 8 * hi;
      ysum[i] += (acc[i] + bc) * rsc[row * R_ + r];
    }
  }
#pragma unroll
  for (int i = 0; i < 8; ++i) {
    const int row = row0 + i + 8 * hi;
    y[(size_t)row * DIM_ + ntile * 16 + n] = ysum[i];
  }
}

// ---------------------------------------------------------------------------
extern "C" void kernel_launch(void* const* d_in, const int* in_sizes, int n_in,
                              void* d_out, int out_size, void* d_ws, size_t ws_size,
                              hipStream_t stream) {
  (void)in_sizes; (void)n_in; (void)out_size; (void)ws_size;

  const float* x    = (const float*)d_in[0];
  const float* rsc  = (const float*)d_in[1];
  const float* Wq   = (const float*)d_in[2];
  const float* bq   = (const float*)d_in[3];
  const float* Wk   = (const float*)d_in[4];
  const float* bk   = (const float*)d_in[5];
  const float* Wv   = (const float*)d_in[6];
  const float* bv   = (const float*)d_in[7];
  const float* Win  = (const float*)d_in[8];
  const float* b_in = (const float*)d_in[9];
  const float* W1   = (const float*)d_in[10];
  const float* b1   = (const float*)d_in[11];
  const float* W2   = (const float*)d_in[12];
  const float* b2   = (const float*)d_in[13];
  float* y = (float*)d_out;

  char* ws = (char*)d_ws;
  size_t off = 0;
  auto alloc = [&](size_t bytes) -> char* {
    char* p = ws + off;
    off = (off + bytes + 255) & ~(size_t)255;
    return p;
  };
  const int KT_D  = DIM_ / 32;   // 16
  const int NT_D  = DIM_ / 16;   // 32
  const int KT_W1 = K1P / 32;    // 5
  const int NT_W1 = 26;
  const int KT_W2 = K2P / 32;    // 13
  const int NT_W2 = 32;

  _Float16* xh    = (_Float16*)alloc((size_t)BN * DIM_ * 2);
  _Float16* wqp   = (_Float16*)alloc((size_t)KT_D * NT_D * TILEH * 2);
  _Float16* wkp   = (_Float16*)alloc((size_t)KT_D * NT_D * TILEH * 2);
  _Float16* wvp   = (_Float16*)alloc((size_t)KT_D * NT_D * TILEH * 2);
  _Float16* winp  = (_Float16*)alloc((size_t)KT_D * 4 * TILEH * 2);
  _Float16* w1p   = (_Float16*)alloc((size_t)R_ * KT_W1 * NT_W1 * TILEH * 2);
  _Float16* w2p   = (_Float16*)alloc((size_t)R_ * KT_W2 * NT_W2 * TILEH * 2);
  _Float16* qh    = (_Float16*)alloc((size_t)BN * DIM_ * 2);
  _Float16* kh    = (_Float16*)alloc((size_t)BN * DIM_ * 2);
  _Float16* vh    = (_Float16*)alloc((size_t)BN * DIM_ * 2);
  _Float16* vpack = (_Float16*)alloc((size_t)8 * G_ * 16 * TILEH * 2);
  _Float16* hbuf  = (_Float16*)alloc((size_t)BN * HROW * 2);
  _Float16* hmid  = (_Float16*)alloc((size_t)BN * MROW * 2);

  // 0) convert x; pack all weights (conversion fused into packing)
  {
    const int cnt = BN * DIM_;
    cvt_f32_f16_kernel<<<(cnt + 1023) / 1024, 256, 0, stream>>>(x, xh, cnt);
  }
  pack_b_f32_kernel<<<KT_D * NT_D, 32, 0, stream>>>(Wq,  wqp,  DIM_, DIM_, KT_D);
  pack_b_f32_kernel<<<KT_D * NT_D, 32, 0, stream>>>(Wk,  wkp,  DIM_, DIM_, KT_D);
  pack_b_f32_kernel<<<KT_D * NT_D, 32, 0, stream>>>(Wv,  wvp,  DIM_, DIM_, KT_D);
  pack_b_f32_kernel<<<KT_D * 4,    32, 0, stream>>>(Win, winp, DIM_, R_ * HD_, KT_D);
  for (int r = 0; r < R_; ++r) {
    pack_b_f32_kernel<<<KT_W1 * NT_W1, 32, 0, stream>>>(
        W1 + (size_t)r * 144 * INTER, w1p + (size_t)r * KT_W1 * NT_W1 * TILEH,
        144, INTER, KT_W1);
    pack_b_f32_kernel<<<KT_W2 * NT_W2, 32, 0, stream>>>(
        W2 + (size_t)r * INTER * DIM_, w2p + (size_t)r * KT_W2 * NT_W2 * TILEH,
        INTER, DIM_, KT_W2);
  }

  // 1) zero concat buffer (pad cols stay zero)
  zero_f16_kernel<<<1024, 256, 0, stream>>>(hbuf, BN * HROW);

  // 2) q/k/v projections; q scaled by HD^-0.5
  {
    dim3 grid(BN / 16, NT_D / 8);
    gemm_bias_f16_kernel<<<grid, 256, 0, stream>>>(xh, wqp, bq, qh, DIM_, DIM_, SCALE_Q);
    gemm_bias_f16_kernel<<<grid, 256, 0, stream>>>(xh, wkp, bk, kh, DIM_, DIM_, 1.0f);
    gemm_bias_f16_kernel<<<grid, 256, 0, stream>>>(xh, wvp, bv, vh, DIM_, DIM_, 1.0f);
  }

  // 3) in-projection -> h[...][128..143]
  inproj_kernel<<<dim3(BN / 16, 1), 256, 0, stream>>>(xh, winp, b_in, hbuf);

  // 4) repack v, then attention -> h[...][0..127]
  pack_v_kernel<<<8 * G_ * 16, 32, 0, stream>>>(vh, vpack);
  attention_kernel<<<dim3(8 * G_, 512 / 16), 32, 0, stream>>>(qh, kh, vpack, hbuf);

  // 5) MLP1 (per rule), ReLU, padded store
  mlp1_kernel<<<dim3(BN / 16, 4, R_), 256, 0, stream>>>(hbuf, w1p, b1, hmid);

  // 6) MLP2 + rule-score mixing -> y
  mlp2_kernel<<<dim3(BN / 16, NT_D / 8), 256, 0, stream>>>(hmid, w2p, b2, rsc, y);

  // 7) second output: r_scores pass-through
  hipMemcpyAsync(y + (size_t)BN * DIM_, rsc, (size_t)BN * R_ * sizeof(float),
                 hipMemcpyDeviceToDevice, stream);
}
